// fusion3_35519379537964
// MI455X (gfx1250) — matrix operand order repaired and checked by reading, not compiled, not run on hardware
//
#include <hip/hip_runtime.h>
#include <hip/hip_bf16.h>

// ---------------- problem constants ----------------
#define BB    2
#define NN    7
#define CC    128
#define HH    120
#define WW    360
#define PP    (HH*WW)          // 43200
#define HID   128
#define GROUPS 16
#define LW    7
#define DIL   2
#define GN_GROUPS 16           // C/8
#define CSIM  176              // 128 + 48
#define KBIG  (898*9)          // 8082
#define KBP   8096             // 253*32, zero-padded
#define KOFF  (176*9)          // 1584
#define KOP   1600             // 50*32, zero-padded
#define NSIM  48

typedef __attribute__((ext_vector_type(8)))  float     v8f;
typedef __attribute__((ext_vector_type(8)))  _Float16  half8;
typedef __attribute__((ext_vector_type(16))) _Float16  half16;
typedef __attribute__((__vector_size__(4 * sizeof(int)))) int v4i;

// gfx1250 async global->LDS DMA (ASYNCcnt-tracked), guarded so compile can't fail
#if defined(__gfx1250__) && __has_builtin(__builtin_amdgcn_global_load_async_to_lds_b128)
#define HAVE_ASYNC_LDS 1
#endif

#ifdef HAVE_ASYNC_LDS
typedef __attribute__((address_space(1))) v4i* gbl_v4i_p;   // printed as "v4i __device__ *"
typedef __attribute__((address_space(3))) v4i* lds_v4i_p;   // printed as "v4i __shared__ *"

__device__ __forceinline__ void async_cp16(const _Float16* g, _Float16* l) {
    __builtin_amdgcn_global_load_async_to_lds_b128(
        (gbl_v4i_p)(void*)g,
        (lds_v4i_p)(void*)l,
        0, 0);
}
__device__ __forceinline__ void wait_async0() {
#if __has_builtin(__builtin_amdgcn_s_wait_asynccnt)
    __builtin_amdgcn_s_wait_asynccnt(0);
#else
    asm volatile("s_wait_asynccnt 0x0" ::: "memory");
#endif
}
#endif

__device__ __forceinline__ v8f wmma_f16f32(const half16 a, const half16 b, v8f c) {
    // D = A(16x32 f16) * B(32x16 f16) + C(16x16 f32)
    return __builtin_amdgcn_wmma_f32_16x16x32_f16(
        /*neg_a=*/false, a, /*neg_b=*/false, b,
        /*c_mod=*/(short)0, c, /*reuse_a=*/false, /*reuse_b=*/false);
}

__device__ __forceinline__ half16 mk_half16(const _Float16* lo, const _Float16* hi) {
    half8 l = *(const half8*)lo;
    half8 h = *(const half8*)hi;
    half16 r;
#pragma unroll
    for (int e = 0; e < 8; ++e) { r[e] = l[e]; r[8 + e] = h[e]; }
    return r;
}

// ---------------- 1) mean over N ----------------
__global__ void mean_kernel(const float* __restrict__ x, float* __restrict__ out) {
    long idx = (long)blockIdx.x * blockDim.x + threadIdx.x;
    const long total = (long)BB * CC * PP;
    if (idx >= total) return;
    int b   = (int)(idx / ((long)CC * PP));
    long rem = idx - (long)b * CC * PP;          // c*P + p
    const float* base = x + (long)b * NN * CC * PP + rem;
    float s = 0.f;
#pragma unroll
    for (int n = 0; n < NN; ++n) s += base[(long)n * CC * PP];
    out[idx] = s * (1.0f / NN);
}

// ---------------- 2) GroupNorm -> lr_sim[0:128] ----------------
__global__ void gn_kernel(const float* __restrict__ fo,
                          const float* __restrict__ gamma,
                          const float* __restrict__ beta,
                          float* __restrict__ lr_sim) {
    __shared__ float s_sum[512];
    __shared__ float s_sq[512];
    int bg = blockIdx.x;              // 0..31
    int b = bg / GN_GROUPS, g = bg % GN_GROUPS;
    int c0 = g * (CC / GN_GROUPS);    // 8 channels
    int tid = threadIdx.x;
    float sum = 0.f, sq = 0.f;
    for (int cc = 0; cc < CC / GN_GROUPS; ++cc) {
        const float* ch = fo + ((long)(b * CC + c0 + cc)) * PP;
        for (int p = tid; p < PP; p += blockDim.x) {
            float v = ch[p]; sum += v; sq += v * v;
        }
    }
    s_sum[tid] = sum; s_sq[tid] = sq;
    __syncthreads();
    for (int s = 256; s > 0; s >>= 1) {
        if (tid < s) { s_sum[tid] += s_sum[tid + s]; s_sq[tid] += s_sq[tid + s]; }
        __syncthreads();
    }
    float inv_n = 1.0f / ((float)(CC / GN_GROUPS) * PP);
    float mu  = s_sum[0] * inv_n;
    float var = s_sq[0] * inv_n - mu * mu;
    float rinv = rsqrtf(var + 1e-5f);
    for (int cc = 0; cc < CC / GN_GROUPS; ++cc) {
        int c = c0 + cc;
        const float* ch = fo + ((long)(b * CC + c)) * PP;
        float* dst = lr_sim + ((long)(b * CSIM + c)) * PP;
        float ga = gamma[c], be = beta[c];
        for (int p = tid; p < PP; p += blockDim.x)
            dst[p] = (ch[p] - mu) * rinv * ga + be;
    }
}

// ---------------- 3) per-pixel channel norm ----------------
__global__ void norm_kernel(const float* __restrict__ lr_sim, float* __restrict__ nrm) {
    int idx = blockIdx.x * blockDim.x + threadIdx.x;
    if (idx >= BB * PP) return;
    int b = idx / PP, p = idx - b * PP;
    const float* base = lr_sim + (long)b * CSIM * PP + p;
    float s = 0.f;
    for (int c = 0; c < CC; ++c) { float v = base[(long)c * PP]; s += v * v; }
    nrm[idx] = fmaxf(sqrtf(s), 1e-8f);
}

// ---------------- 4) 48-tap cosine similarity -> lr_sim[128:176] ----------------
__global__ void sim_kernel(const float* __restrict__ lr_sim_in,
                           const float* __restrict__ nrm,
                           float* __restrict__ lr_sim_out) {
    int idx = blockIdx.x * blockDim.x + threadIdx.x;
    if (idx >= BB * PP) return;
    int b = idx / PP, p = idx - b * PP;
    int h = p / WW, w = p - h * WW;
    int np[NSIM];
    {
        int t = 0;
#pragma unroll
        for (int i = 0; i < LW; ++i)
#pragma unroll
            for (int j = 0; j < LW; ++j) {
                if (i == LW / 2 && j == LW / 2) continue;
                int hh = h + (i - LW / 2) * DIL;
                int ww = w + (j - LW / 2) * DIL;
                np[t++] = (hh >= 0 && hh < HH && ww >= 0 && ww < WW) ? (hh * WW + ww) : -1;
            }
    }
    float acc[NSIM];
#pragma unroll
    for (int t = 0; t < NSIM; ++t) acc[t] = 0.f;
    const float* base = lr_sim_in + (long)b * CSIM * PP;
    for (int c = 0; c < CC; ++c) {
        const float* ch = base + (long)c * PP;
        float xc = ch[p];
#pragma unroll
        for (int t = 0; t < NSIM; ++t)
            if (np[t] >= 0) acc[t] += xc * ch[np[t]];
    }
    float cn = nrm[(long)b * PP + p];
#pragma unroll
    for (int t = 0; t < NSIM; ++t) {
        float v = 0.f;
        if (np[t] >= 0) {
            float pn = nrm[(long)b * PP + np[t]];
            v = acc[t] / (cn * pn);
        }
        lr_sim_out[((long)(b * CSIM + CC + t)) * PP + p] = v;
    }
}

// ---------------- weight pre-conversion to f16, zero-padded K ----------------
__global__ void cvt_wbig_kernel(const float* __restrict__ cw, _Float16* __restrict__ wh) {
    long i = (long)blockIdx.x * blockDim.x + threadIdx.x;
    const long total = (long)HID * KBP;
    if (i >= total) return;
    int oc = (int)(i / KBP);
    int k  = (int)(i - (long)oc * KBP);
    __builtin_prefetch(cw + (long)oc * KBIG + k + 256, 0, 1);
    wh[i] = (k < KBIG) ? (_Float16)cw[(long)oc * KBIG + k] : (_Float16)0.0f;
}

__global__ void cvt_woff_kernel(const float* __restrict__ w0, const float* __restrict__ w1,
                                const float* __restrict__ w2, _Float16* __restrict__ wh) {
    long i = (long)blockIdx.x * blockDim.x + threadIdx.x;
    const long total = (long)64 * KOP;
    if (i >= total) return;
    int oc = (int)(i / KOP);
    int k  = (int)(i - (long)oc * KOP);
    float v = 0.f;
    if (k < KOFF) {
        if (oc < 32)      v = w0[(long)oc * KOFF + k];
        else if (oc < 48) v = w1[(long)(oc - 32) * KOFF + k];
        else if (oc < 56) v = w2[(long)(oc - 48) * KOFF + k];
    }
    wh[i] = (_Float16)v;
}

// ---------------- 5) big conv (898ch in, 128ch out, 3x3, relu) via WMMA ----------------
// implicit GEMM: M = 64 pixels/block, N = 128 oc (8 waves x 16), K in steps of 32
#define AS 40   // LDS row stride in halfs (80B: 16B aligned, bank-rotating)
__global__ void __launch_bounds__(256)
conv_big_kernel(const float* __restrict__ x, const _Float16* __restrict__ wh,
                const float* __restrict__ cb, float* __restrict__ feat) {
    __shared__ __align__(16) _Float16 Asmem[64 * AS];
    __shared__ __align__(16) _Float16 Bsmem[128 * AS];
    int bid = blockIdx.x;
    int b = bid / (PP / 64);
    int p0 = (bid - b * (PP / 64)) * 64;
    int tid = threadIdx.x;
    int wave = tid >> 5, lane = tid & 31;
    int half = lane >> 4, ln = lane & 15;

    v8f acc[4];
#pragma unroll
    for (int m = 0; m < 4; ++m) acc[m] = (v8f)0.0f;

    const int nsteps = KBP / 32;   // 253
    for (int ks = 0; ks < nsteps; ++ks) {
        int k0 = ks * 32;
        // stage B^T: 128 oc x 32 halfs = 512 x 16B chunks (pure byte move, 16B aligned)
#ifdef HAVE_ASYNC_LDS
        for (int e = tid; e < 512; e += 256) {
            int oc = e >> 2, q = (e & 3) * 8;
            async_cp16(wh + (long)oc * KBP + k0 + q, &Bsmem[oc * AS + q]);
        }
#else
        for (int e = tid; e < 512; e += 256) {
            int oc = e >> 2, q = (e & 3) * 8;
            *(half8*)&Bsmem[oc * AS + q] = *(const half8*)(wh + (long)oc * KBP + k0 + q);
        }
#endif
        // stage A: 64 pixels x 32 k  (f32 -> f16 gather, overlaps with B DMA)
        for (int e = tid; e < 64 * 32; e += 256) {
            int m = e >> 5, kk = e & 31;
            int k = k0 + kk;
            float v = 0.f;
            if (k < KBIG) {
                int ic = k / 9;
                int t = k - ic * 9;
                int ky = t / 3, kx = t - ky * 3;
                int p = p0 + m;
                int h = p / WW, w = p - h * WW;
                int hh = h + ky - 1, ww = w + kx - 1;
                if (hh >= 0 && hh < HH && ww >= 0 && ww < WW) {
                    if (ic < NN * CC) {
                        int n = ic >> 7, c = ic & 127;
                        v = x[(((long)(b * NN + n) * CC + c)) * PP + hh * WW + ww];
                    } else if (ic == NN * CC) {
                        v = (float)ww * (2.0f / (WW - 1)) - 1.0f;   // coord gx
                    } else {
                        v = (float)hh * (2.0f / (HH - 1)) - 1.0f;   // coord gy
                    }
                }
            }
            Asmem[m * AS + kk] = (_Float16)v;
        }
#ifdef HAVE_ASYNC_LDS
        wait_async0();
#endif
        __syncthreads();
        // B fragment: lane holds column N=ln, K = 16*half .. +15
        half16 bf = mk_half16(&Bsmem[(wave * 16 + ln) * AS + 16 * half],
                              &Bsmem[(wave * 16 + ln) * AS + 16 * half + 8]);
#pragma unroll
        for (int ms = 0; ms < 4; ++ms) {
            // A fragment: lane holds row M=ln, K = 16*half .. +15
            const _Float16* arow = &Asmem[(ms * 16 + ln) * AS];
            half16 af = mk_half16(arow + 16 * half, arow + 16 * half + 8);
            acc[ms] = wmma_f16f32(af, bf, acc[ms]);
        }
        __syncthreads();
    }
    // writeback: C/D layout -> VGPR r : M = r + 8*half, N = ln
    int oc = wave * 16 + ln;
    float bias = cb[oc];
#pragma unroll
    for (int ms = 0; ms < 4; ++ms)
#pragma unroll
        for (int r = 0; r < 8; ++r) {
            int p = p0 + ms * 16 + r + 8 * half;
            feat[((long)(b * HID + oc)) * PP + p] = fmaxf(acc[ms][r] + bias, 0.f);
        }
}

// ---------------- 6) fused offset convs (176ch -> 32+16+8 padded to 64) via WMMA ----------------
__global__ void __launch_bounds__(128)
conv_off_kernel(const float* __restrict__ lr_sim, const _Float16* __restrict__ wh,
                const float* __restrict__ b0, const float* __restrict__ b1,
                const float* __restrict__ b2, float* __restrict__ offs) {
    __shared__ __align__(16) _Float16 Asmem[64 * AS];
    __shared__ __align__(16) _Float16 Bsmem[64 * AS];
    int bid = blockIdx.x;
    int b = bid / (PP / 64);
    int p0 = (bid - b * (PP / 64)) * 64;
    int tid = threadIdx.x;
    int wave = tid >> 5, lane = tid & 31;
    int half = lane >> 4, ln = lane & 15;

    v8f acc[4];
#pragma unroll
    for (int m = 0; m < 4; ++m) acc[m] = (v8f)0.0f;

    const int nsteps = KOP / 32;   // 50
    for (int ks = 0; ks < nsteps; ++ks) {
        int k0 = ks * 32;
        // stage B^T: 64 oc x 32 halfs = 256 x 16B chunks
#ifdef HAVE_ASYNC_LDS
        for (int e = tid; e < 256; e += 128) {
            int oc = e >> 2, q = (e & 3) * 8;
            async_cp16(wh + (long)oc * KOP + k0 + q, &Bsmem[oc * AS + q]);
        }
#else
        for (int e = tid; e < 256; e += 128) {
            int oc = e >> 2, q = (e & 3) * 8;
            *(half8*)&Bsmem[oc * AS + q] = *(const half8*)(wh + (long)oc * KOP + k0 + q);
        }
#endif
        // stage A from lr_sim (f32 -> f16)
        for (int e = tid; e < 64 * 32; e += 128) {
            int m = e >> 5, kk = e & 31;
            int k = k0 + kk;
            float v = 0.f;
            if (k < KOFF) {
                int ic = k / 9;
                int t = k - ic * 9;
                int ky = t / 3, kx = t - ky * 3;
                int p = p0 + m;
                int h = p / WW, w = p - h * WW;
                int hh = h + ky - 1, ww = w + kx - 1;
                if (hh >= 0 && hh < HH && ww >= 0 && ww < WW)
                    v = lr_sim[((long)(b * CSIM + ic)) * PP + hh * WW + ww];
            }
            Asmem[m * AS + kk] = (_Float16)v;
        }
#ifdef HAVE_ASYNC_LDS
        wait_async0();
#endif
        __syncthreads();
        half16 bf = mk_half16(&Bsmem[(wave * 16 + ln) * AS + 16 * half],
                              &Bsmem[(wave * 16 + ln) * AS + 16 * half + 8]);
#pragma unroll
        for (int ms = 0; ms < 4; ++ms) {
            const _Float16* arow = &Asmem[(ms * 16 + ln) * AS];
            half16 af = mk_half16(arow + 16 * half, arow + 16 * half + 8);
            acc[ms] = wmma_f16f32(af, bf, acc[ms]);
        }
        __syncthreads();
    }
    int oc = wave * 16 + ln;
    if (oc < 56) {
        float bias = (oc < 32) ? b0[oc] : (oc < 48) ? b1[oc - 32] : b2[oc - 48];
#pragma unroll
        for (int ms = 0; ms < 4; ++ms)
#pragma unroll
            for (int r = 0; r < 8; ++r) {
                int p = p0 + ms * 16 + r + 8 * half;
                offs[((long)(b * 56 + oc)) * PP + p] = acc[ms][r] + bias;
            }
    }
}

// ---------------- 7) 3-way deformable bilinear gather + average ----------------
// grid_sample algebra collapses: ix = w + offx, iy = h + offy (border clamp)
__device__ __forceinline__ float bilin(const float* __restrict__ ch, float ix, float iy) {
    float x0f = floorf(ix), y0f = floorf(iy);
    float wx = ix - x0f, wy = iy - y0f;
    int x0 = (int)fminf(fmaxf(x0f,        0.f), (float)(WW - 1));
    int x1 = (int)fminf(fmaxf(x0f + 1.f,  0.f), (float)(WW - 1));
    int y0 = (int)fminf(fmaxf(y0f,        0.f), (float)(HH - 1));
    int y1 = (int)fminf(fmaxf(y0f + 1.f,  0.f), (float)(HH - 1));
    float v00 = ch[y0 * WW + x0], v01 = ch[y0 * WW + x1];
    float v10 = ch[y1 * WW + x0], v11 = ch[y1 * WW + x1];
    return v00 * (1.f - wx) * (1.f - wy) + v01 * wx * (1.f - wy)
         + v10 * (1.f - wx) * wy        + v11 * wx * wy;
}

__global__ void sample_kernel(const float* __restrict__ feat,
                              const float* __restrict__ offs,
                              float* __restrict__ out) {
    long idx = (long)blockIdx.x * blockDim.x + threadIdx.x;
    const long total = (long)BB * CC * PP;
    if (idx >= total) return;
    int b = (int)(idx / ((long)CC * PP));
    long r = idx - (long)b * CC * PP;
    int c = (int)(r / PP);
    int p = (int)(r - (long)c * PP);
    int h = p / WW, w = p - h * WW;
    const float* ch = feat + ((long)(b * HID + c)) * PP;
    const float* ob = offs + (long)b * 56 * PP + p;
    int g0 = c >> 3;   // 16 groups, 8 ch each
    int g1 = c >> 4;   // 8 groups
    int g2 = c >> 5;   // 4 groups
    float r0 = bilin(ch, (float)w + ob[(long)g0 * PP],        (float)h + ob[(long)(16 + g0) * PP]);
    float r1 = bilin(ch, (float)w + ob[(long)(32 + g1) * PP], (float)h + ob[(long)(40 + g1) * PP]);
    float r2 = bilin(ch, (float)w + ob[(long)(48 + g2) * PP], (float)h + ob[(long)(52 + g2) * PP]);
    out[idx] = (r0 + r1 + r2) * (1.0f / 3.0f);
}

// ---------------- launch ----------------
extern "C" void kernel_launch(void* const* d_in, const int* in_sizes, int n_in,
                              void* d_out, int out_size, void* d_ws, size_t ws_size,
                              hipStream_t stream) {
    const float* x     = (const float*)d_in[0];
    const float* cw    = (const float*)d_in[1];
    const float* cb    = (const float*)d_in[2];
    const float* gamma = (const float*)d_in[3];
    const float* beta  = (const float*)d_in[4];
    const float* ow0   = (const float*)d_in[5];
    const float* obi0  = (const float*)d_in[6];
    const float* ow1   = (const float*)d_in[7];
    const float* obi1  = (const float*)d_in[8];
    const float* ow2   = (const float*)d_in[9];
    const float* obi2  = (const float*)d_in[10];
    float* out = (float*)d_out;

    float* ws      = (float*)d_ws;
    float* for_off = ws;                               // B*C*P
    float* lr_sim  = for_off + (long)BB * CC * PP;     // B*176*P
    float* nrm     = lr_sim  + (long)BB * CSIM * PP;   // B*P
    float* feat    = nrm     + (long)BB * PP;          // B*128*P
    float* offs    = feat    + (long)BB * HID * PP;    // B*56*P
    _Float16* wbig_h = (_Float16*)(offs + (long)BB * 56 * PP);  // 128*KBP halfs (16B aligned)
    _Float16* woff_h = wbig_h + (long)HID * KBP;                 // 64*KOP halfs

    const long nelem = (long)BB * CC * PP;

    mean_kernel<<<(unsigned)((nelem + 255) / 256), 256, 0, stream>>>(x, for_off);
    gn_kernel<<<BB * GN_GROUPS, 512, 0, stream>>>(for_off, gamma, beta, lr_sim);
    norm_kernel<<<(BB * PP + 255) / 256, 256, 0, stream>>>(lr_sim, nrm);
    sim_kernel<<<(BB * PP + 255) / 256, 256, 0, stream>>>(lr_sim, nrm, lr_sim);

    cvt_wbig_kernel<<<(unsigned)(((long)HID * KBP + 255) / 256), 256, 0, stream>>>(cw, wbig_h);
    cvt_woff_kernel<<<(unsigned)(((long)64 * KOP + 255) / 256), 256, 0, stream>>>(ow0, ow1, ow2, woff_h);

    conv_big_kernel<<<BB * (PP / 64), 256, 0, stream>>>(x, wbig_h, cb, feat);
    conv_off_kernel<<<BB * (PP / 64), 128, 0, stream>>>(lr_sim, woff_h, obi0, obi1, obi2, offs);
    sample_kernel<<<(unsigned)((nelem + 255) / 256), 256, 0, stream>>>(feat, offs, out);
}
